// ResMRConvWithEdgeFeats_59150289600864
// MI455X (gfx1250) — matrix operand order repaired and checked by reading, not compiled, not run on hardware
//
#include <hip/hip_runtime.h>
#include <hip/hip_bf16.h>

// ---------------------------------------------------------------------------
// ResMRConv with edge features, CDNA5 (gfx1250) WMMA implementation.
//   out = x + relu([x ; segmax] @ W_mlp + b_mlp)
//   segmax = segment_max over dst of (diffs + relu([diffs ; e_feat] @ W_edge + b_edge))
// bf16 A/B fragments, f32 accumulation via v_wmma_f32_16x16x32_bf16.
// Scatter-max uses branch-free "sortable float" keys + global_atomic_max_u32.
// 64 rows per block (4 M-tiles x 8 N-tile waves) to amortize B-fragment loads.
// ---------------------------------------------------------------------------

typedef __attribute__((ext_vector_type(16))) __bf16 v16bf;
typedef __attribute__((ext_vector_type(8)))  __bf16 v8bf;
typedef __attribute__((ext_vector_type(8)))  float  v8f;

#define WIDTH 128
#define NFEAT 64
#define KEDGE 192   // WIDTH + NFEAT
#define KMLP  256   // 2*WIDTH
#define KT_EDGE 6   // 192/32
#define KT_MLP  8   // 256/32
#define MT 4        // M-tiles (of 16 rows) per block
#define ROWS (MT * 16)
#define FRAG_ELEMS 512                      // one 32x16 bf16 B fragment
#define NE_PACK (KT_EDGE * 8 * FRAG_ELEMS)  // 24576
#define NM_PACK (KT_MLP  * 8 * FRAG_ELEMS)  // 32768
#define ENC_NEG_INF 0x007FFFFFu             // sortable-key encoding of -inf

// Monotonic float <-> uint key: unsigned order == float order (all signs).
__device__ __forceinline__ unsigned enc_f32(float f) {
  unsigned u = __float_as_uint(f);
  return (u & 0x80000000u) ? ~u : (u | 0x80000000u);
}
__device__ __forceinline__ float dec_f32(unsigned k) {
  unsigned u = (k & 0x80000000u) ? (k & 0x7fffffffu) : ~k;
  return __uint_as_float(u);
}

// ---------------------------------------------------------------------------
// Pack W (K x 128, row-major f32) into bf16 WMMA-B fragment order.
// B layout (16-bit, 32x16): lanes 0-15 hold K = {0..7, 16..23}, N = lane;
//                           lanes 16-31 hold K = {8..15, 24..31}, N = lane-16.
// ---------------------------------------------------------------------------
__global__ __launch_bounds__(256) void pack_weights_kernel(
    const float* __restrict__ We, const float* __restrict__ Wm,
    unsigned short* __restrict__ pe, unsigned short* __restrict__ pm) {
  int idx = blockIdx.x * 256 + threadIdx.x;
  if (idx >= NE_PACK + NM_PACK) return;
  const float* W;
  unsigned short* dst;
  int li;
  if (idx < NE_PACK) { W = We; dst = pe; li = idx; }
  else               { W = Wm; dst = pm; li = idx - NE_PACK; }
  int frag = li >> 9;          // /512
  int rem  = li & 511;
  int lane = rem >> 4;         // 0..31
  int q    = rem & 15;         // element within lane's 16
  int v    = q >> 1;
  int h    = q & 1;
  int kt   = frag >> 3;
  int nt   = frag & 7;
  int n    = lane & 15;
  int k    = ((q & 8) ? 16 : 0) + ((lane >> 4) ? 8 : 0) + ((2 * v + h) & 7);
  float val = W[(kt * 32 + k) * WIDTH + nt * 16 + n];
  __bf16 bv = (__bf16)val;
  dst[li] = __builtin_bit_cast(unsigned short, bv);
}

// Fill maxes with encoded -inf (segment_max identity under u32-max).
__global__ __launch_bounds__(256) void init_maxes_kernel(unsigned* __restrict__ p, int n) {
  int idx = blockIdx.x * 256 + threadIdx.x;
  if (idx < n) p[idx] = ENC_NEG_INF;
}

// Build v16bf A fragment from LDS row-major bf16 tile.
// A layout (16-bit, 16x32): lanes 0-15: M=lane, K = {0..7, 16..23};
//                           lanes 16-31: M=lane-16, K = {8..15, 24..31}.
__device__ __forceinline__ v16bf load_a_frag(const __bf16* smA, int lda, int kt,
                                             int mtile, int lane) {
  int mrow  = (lane & 15) + mtile * 16;
  int khalf = (lane >> 4) ? 8 : 0;
  const __bf16* ap = smA + mrow * lda + kt * 32 + khalf;
  v8bf a0 = *(const v8bf*)ap;          // K 0..7 (or 8..15)
  v8bf a1 = *(const v8bf*)(ap + 16);   // K 16..23 (or 24..31)
  v16bf a;
#pragma unroll
  for (int i = 0; i < 8; ++i) { a[i] = a0[i]; a[i + 8] = a1[i]; }
  return a;
}

// ---------------------------------------------------------------------------
// Edge kernel: 64 edges / block, 8 waves (one 16-col N-tile each, 4 M-tiles).
// ---------------------------------------------------------------------------
#define LDA_E 200   // 192 padded; keeps 16-B alignment, spreads LDS banks
__global__ __launch_bounds__(256) void edge_kernel(
    const float* __restrict__ x, const int* __restrict__ e,
    const float* __restrict__ e_feat, const unsigned short* __restrict__ wpack,
    const float* __restrict__ b_edge, unsigned* __restrict__ maxes, int E) {
  __shared__ __bf16 smA[ROWS * LDA_E];
  __shared__ int ssrc[ROWS];
  __shared__ int sdst[ROWS];

  int tid  = threadIdx.x;
  int lane = tid & 31;
  int wv   = tid >> 5;          // N-tile 0..7
  long eBase = (long)blockIdx.x * ROWS;

  if (tid < ROWS) {
    long eidx = eBase + tid;
    if (eidx >= E) eidx = E - 1;          // duplicate edge: idempotent under max
    ssrc[tid] = e[eidx];
    sdst[tid] = e[(long)E + eidx];
  }
  __syncthreads();

  // Stage A = [diffs(128) ; e_feat(64)] as bf16, row-major with pad.
  for (int i = tid; i < ROWS * KEDGE; i += 256) {
    int row = i / KEDGE;
    int col = i - row * KEDGE;
    long eidx = eBase + row;
    if (eidx >= E) eidx = E - 1;
    float v;
    if (col < WIDTH)
      v = x[(long)sdst[row] * WIDTH + col] - x[(long)ssrc[row] * WIDTH + col];
    else
      v = e_feat[eidx * NFEAT + (col - WIDTH)];
    smA[row * LDA_E + col] = (__bf16)v;
  }
  __syncthreads();

  const __bf16* wp = (const __bf16*)wpack;
  v8f acc[MT];
#pragma unroll
  for (int mt = 0; mt < MT; ++mt) acc[mt] = {};

#pragma unroll
  for (int kt = 0; kt < KT_EDGE; ++kt) {
    v16bf b = *(const v16bf*)(wp + (kt * 8 + wv) * FRAG_ELEMS + lane * 16);
#pragma unroll
    for (int mt = 0; mt < MT; ++mt) {
      v16bf a = load_a_frag(smA, LDA_E, kt, mt, lane);
      acc[mt] = __builtin_amdgcn_wmma_f32_16x16x32_bf16(
          false, a, false, b, (short)0, acc[mt], false, false);
    }
  }

  // Epilogue: bias + relu + diffs residual, branch-free scatter-max.
  int n    = (lane & 15) + wv * 16;   // output column 0..127
  float bias = b_edge[n];
  int mAdd = (lane >> 4) ? 8 : 0;
#pragma unroll
  for (int mt = 0; mt < MT; ++mt) {
#pragma unroll
    for (int r = 0; r < 8; ++r) {
      int m = mt * 16 + r + mAdd;     // edge row within block
      float v = fmaxf(acc[mt][r] + bias, 0.0f);
      v += (float)smA[m * LDA_E + n];  // + diffs (n < 128)
      unsigned* addr = maxes + (long)sdst[m] * WIDTH + n;
      atomicMax(addr, enc_f32(v));     // single unconditional u32 atomic
    }
  }
}

// ---------------------------------------------------------------------------
// Node kernel: 64 nodes / block, 8 waves, 4 M-tiles. K = 256 = [x ; segmax].
// ---------------------------------------------------------------------------
#define LDA_N 264   // 256 padded
__global__ __launch_bounds__(256) void node_kernel(
    const float* __restrict__ x, const unsigned* __restrict__ maxes,
    const unsigned short* __restrict__ wpack, const float* __restrict__ b_mlp,
    float* __restrict__ out, int N) {
  __shared__ __bf16 smA[ROWS * LDA_N];

  int tid  = threadIdx.x;
  int lane = tid & 31;
  int wv   = tid >> 5;
  long base = (long)blockIdx.x * ROWS;

  for (int i = tid; i < ROWS * KMLP; i += 256) {
    int row = i >> 8;        // /256
    int col = i & 255;
    long node = base + row;
    if (node >= N) node = N - 1;
    float v;
    if (col < WIDTH) {
      v = x[node * WIDTH + col];
    } else {
      v = dec_f32(maxes[node * WIDTH + (col - WIDTH)]);
      if (v < -3.0e38f) v = 0.0f;   // empty segment (-inf) -> 0
    }
    smA[row * LDA_N + col] = (__bf16)v;
  }
  __syncthreads();

  const __bf16* wp = (const __bf16*)wpack;
  v8f acc[MT];
#pragma unroll
  for (int mt = 0; mt < MT; ++mt) acc[mt] = {};

#pragma unroll
  for (int kt = 0; kt < KT_MLP; ++kt) {
    v16bf b = *(const v16bf*)(wp + (kt * 8 + wv) * FRAG_ELEMS + lane * 16);
#pragma unroll
    for (int mt = 0; mt < MT; ++mt) {
      v16bf a = load_a_frag(smA, LDA_N, kt, mt, lane);
      acc[mt] = __builtin_amdgcn_wmma_f32_16x16x32_bf16(
          false, a, false, b, (short)0, acc[mt], false, false);
    }
  }

  int n = (lane & 15) + wv * 16;
  float bias = b_mlp[n];
  int mAdd = (lane >> 4) ? 8 : 0;
#pragma unroll
  for (int mt = 0; mt < MT; ++mt) {
#pragma unroll
    for (int r = 0; r < 8; ++r) {
      long node = base + mt * 16 + r + mAdd;
      if (node < N) {
        float v = fmaxf(acc[mt][r] + bias, 0.0f);
        out[node * WIDTH + n] = x[node * WIDTH + n] + v;
      }
    }
  }
}

// ---------------------------------------------------------------------------
extern "C" void kernel_launch(void* const* d_in, const int* in_sizes, int n_in,
                              void* d_out, int out_size, void* d_ws, size_t ws_size,
                              hipStream_t stream) {
  const float* x      = (const float*)d_in[0];
  const int*   e      = (const int*)d_in[1];
  const float* e_feat = (const float*)d_in[2];
  const float* W_edge = (const float*)d_in[3];
  const float* b_edge = (const float*)d_in[4];
  const float* W_mlp  = (const float*)d_in[5];
  const float* b_mlp  = (const float*)d_in[6];
  float* out = (float*)d_out;

  int N = in_sizes[0] / WIDTH;
  int E = in_sizes[1] / 2;

  // Workspace layout: [W_edge packed bf16][W_mlp packed bf16][maxes keys N*128]
  char* ws = (char*)d_ws;
  unsigned short* wpe = (unsigned short*)ws;                       // 49152 B
  unsigned short* wpm = (unsigned short*)(ws + NE_PACK * 2);       // 65536 B
  unsigned* maxes = (unsigned*)(ws + NE_PACK * 2 + NM_PACK * 2);   // N*128*4 B

  int packTot = NE_PACK + NM_PACK;
  pack_weights_kernel<<<(packTot + 255) / 256, 256, 0, stream>>>(W_edge, W_mlp, wpe, wpm);

  int maxElems = N * WIDTH;
  init_maxes_kernel<<<(maxElems + 255) / 256, 256, 0, stream>>>(maxes, maxElems);

  int edgeBlocks = (E + ROWS - 1) / ROWS;
  edge_kernel<<<edgeBlocks, 256, 0, stream>>>(x, e, e_feat, wpe, b_edge, maxes, E);

  int nodeBlocks = (N + ROWS - 1) / ROWS;
  node_kernel<<<nodeBlocks, 256, 0, stream>>>(x, maxes, wpm, b_mlp, out, N);
}
